// Enc_54537494725140
// MI455X (gfx1250) — matrix operand order repaired and checked by reading, not compiled, not run on hardware
//
#include <hip/hip_runtime.h>

typedef __attribute__((ext_vector_type(16))) _Float16 v16h;
typedef __attribute__((ext_vector_type(8)))  float    v8f;

#define B_DIM  2
#define NA     1500
#define NC     64
#define FD     128
#define F3     384
#define NTILE  ((NA + 15) / 16)   // 94 atom tiles of 16
#define CUTOFF 5.0f

// f16 A-matrix 16x32 fragment layout (ISA 7.12.2):
// lane L: M=L%16, h2=L/16; vector elem e (v=e/2, par=e&1):
//   K = (v<4 ? 2v : 16+2(v-4)) + 8*h2 + par
__device__ __forceinline__ int a_klocal(int e, int h2) {
    int v = e >> 1, par = e & 1;
    return (v < 4 ? 2 * v : 16 + 2 * (v - 4)) + 8 * h2 + par;
}

// --------------------------------------------------------------------------
// Kernel 1: phi1[b,a,f] = (silu(h@W1+b1)@W2 + b2)[3f+1], phi2 = [...][3f+2]
// One block per (b, 16-atom tile); 8 waves, wave w owns columns [16w,16w+16).
// --------------------------------------------------------------------------
__global__ void __launch_bounds__(256)
phi_kernel(const float* __restrict__ hin,
           const float* __restrict__ W1, const float* __restrict__ b1,
           const float* __restrict__ W2, const float* __restrict__ b2,
           float* __restrict__ phi1, float* __restrict__ phi2) {
    __shared__ __align__(32) _Float16 aFrag[4][32][16];   // 4 K-chunks of A frags

    const int tile = blockIdx.x % NTILE;
    const int b    = blockIdx.x / NTILE;
    const int a0   = tile * 16;
    const int tid  = threadIdx.x;
    const int lane = tid & 31;
    const int wid  = tid >> 5;
    const int nloc = lane & 15;
    const int h2   = lane >> 4;
    const int col  = 16 * wid + nloc;     // this lane's output column (0..127)

    // Stage h tile (16 atoms x 128) into fragment-ordered f16 LDS.
    {
        const int chunk = tid >> 6, rem = tid & 63;
        const int flane = rem >> 1, ehalf = rem & 1;
        const int frow = flane & 15, fh2 = flane >> 4;
        const int atom = a0 + frow;
        #pragma unroll
        for (int i = 0; i < 8; ++i) {
            int e = ehalf * 8 + i;
            int k = chunk * 32 + a_klocal(e, fh2);
            float v = (atom < NA) ? hin[((size_t)b * NA + atom) * FD + k] : 0.0f;
            aFrag[chunk][flane][e] = (_Float16)v;
        }
    }
    __syncthreads();

    // GEMM1: t = silu(h @ W1 + b1)
    v8f acc = {};
    #pragma unroll
    for (int c = 0; c < 4; ++c) {
        v16h aF = *(const v16h*)(&aFrag[c][lane][0]);
        v16h bF;
        #pragma unroll
        for (int e = 0; e < 16; ++e) {          // B 32x16: same K order as A frags
            int k = 32 * c + 8 * h2 + (e < 8 ? e : e + 8);
            bF[e] = (_Float16)W1[k * FD + col];
        }
        acc = __builtin_amdgcn_wmma_f32_16x16x32_f16(false, aF, false, bF,
                                                     (short)0, acc, false, false);
    }
    float tv[8];
    #pragma unroll
    for (int r = 0; r < 8; ++r) {
        float x = acc[r] + b1[col];
        tv[r] = x / (1.0f + __expf(-x));        // silu
    }
    __syncthreads();                            // all reads of aFrag done

    // Re-layout silu output (D layout) back into A-fragment order for GEMM2.
    #pragma unroll
    for (int r = 0; r < 8; ++r) {
        int row = r + 8 * h2;                   // atom row
        int k = col;                            // K index of GEMM2
        int chunk = k >> 5, kl = k & 31;
        int th2 = (kl >> 3) & 1;
        int rest = kl - 8 * th2;
        int v = (rest < 8) ? (rest >> 1) : (4 + ((rest - 16) >> 1));
        int e = 2 * v + (rest & 1);
        aFrag[chunk][row + 16 * th2][e] = (_Float16)tv[r];
    }
    __syncthreads();

    // GEMM2: phi1/phi2 = columns (3f+1, 3f+2) of t @ W2 + b2
    v8f acc1 = {}, acc2 = {};
    #pragma unroll
    for (int c = 0; c < 4; ++c) {
        v16h aF = *(const v16h*)(&aFrag[c][lane][0]);
        v16h bF1, bF2;
        #pragma unroll
        for (int e = 0; e < 16; ++e) {
            int k = 32 * c + 8 * h2 + (e < 8 ? e : e + 8);
            bF1[e] = (_Float16)W2[k * F3 + 3 * col + 1];
            bF2[e] = (_Float16)W2[k * F3 + 3 * col + 2];
        }
        acc1 = __builtin_amdgcn_wmma_f32_16x16x32_f16(false, aF, false, bF1,
                                                      (short)0, acc1, false, false);
        acc2 = __builtin_amdgcn_wmma_f32_16x16x32_f16(false, aF, false, bF2,
                                                      (short)0, acc2, false, false);
    }
    #pragma unroll
    for (int r = 0; r < 8; ++r) {
        int atom = a0 + r + 8 * h2;
        if (atom < NA) {
            size_t idx = ((size_t)b * NA + atom) * FD + col;
            phi1[idx] = acc1[r] + b2[3 * col + 1];
            phi2[idx] = acc2[r] + b2[3 * col + 2];
        }
    }
}

// --------------------------------------------------------------------------
// Kernel 2: per (b,c): E[a,g]=exp(-(d-off_g)^2); w{1,2}=E@Wf[:,3f+{1,2}];
// dH[f] += w1*phi1; dV[f,:] += w2*phi2*unit.  One block per (b,c), 8 waves.
// --------------------------------------------------------------------------
__global__ void __launch_bounds__(256)
agg_kernel(const float* __restrict__ Hc,
           const float* __restrict__ cg_xyz,
           const float* __restrict__ xyz,
           const float* __restrict__ Wf, const float* __restrict__ bf,
           const float* __restrict__ phi1, const float* __restrict__ phi2,
           float* __restrict__ out) {
    __shared__ __align__(32) _Float16 aFrag[4][32][16];   // RBF E fragments
    __shared__ float uvec[16][4];                         // unit xyz + d per atom

    const int c    = blockIdx.x % NC;
    const int b    = blockIdx.x / NC;
    const int tid  = threadIdx.x;
    const int lane = tid & 31;
    const int wid  = tid >> 5;
    const int nloc = lane & 15;
    const int h2   = lane >> 4;
    const int fcol = 16 * wid + nloc;         // this lane's f column

    const float cx = cg_xyz[(b * NC + c) * 3 + 0];
    const float cy = cg_xyz[(b * NC + c) * 3 + 1];
    const float cz = cg_xyz[(b * NC + c) * 3 + 2];

    // Resident B fragments of Wf columns 3f+1 (dH path) and 3f+2 (dV path):
    // loaded once, reused for all 94 atom tiles.
    v16h bW1[4], bW2[4];
    #pragma unroll
    for (int cc = 0; cc < 4; ++cc) {
        #pragma unroll
        for (int e = 0; e < 16; ++e) {
            int k = 32 * cc + 8 * h2 + (e < 8 ? e : e + 8);
            bW1[cc][e] = (_Float16)Wf[k * F3 + 3 * fcol + 1];
            bW2[cc][e] = (_Float16)Wf[k * F3 + 3 * fcol + 2];
        }
    }
    const float bf1 = bf[3 * fcol + 1];
    const float bf2 = bf[3 * fcol + 2];

    float sH = 0.f, sVx = 0.f, sVy = 0.f, sVz = 0.f;

    const int chunk = tid >> 6, rem = tid & 63;
    const int flane = rem >> 1, ehalf = rem & 1;
    const int frow = flane & 15, fh2 = flane >> 4;
    const float offStep = CUTOFF / (float)(FD - 1);

    for (int t0 = 0; t0 < NTILE; ++t0) {
        const int a0 = t0 * 16;
        // Build Gaussian RBF fragments E[a,g] = exp(-(d - g*step)^2) in LDS.
        {
            int atom = a0 + frow;
            int ai = (atom < NA) ? atom : (NA - 1);
            float dx = xyz[((size_t)b * NA + ai) * 3 + 0] - cx;
            float dy = xyz[((size_t)b * NA + ai) * 3 + 1] - cy;
            float dz = xyz[((size_t)b * NA + ai) * 3 + 2] - cz;
            float d = sqrtf(dx * dx + dy * dy + dz * dz);
            if (tid < 16) {                     // unit vectors for this tile
                float inv = 1.0f / d;
                uvec[tid][0] = dx * inv;
                uvec[tid][1] = dy * inv;
                uvec[tid][2] = dz * inv;
                uvec[tid][3] = d;
            }
            #pragma unroll
            for (int i = 0; i < 8; ++i) {
                int e = ehalf * 8 + i;
                int g = chunk * 32 + a_klocal(e, fh2);
                float diff = d - (float)g * offStep;
                float Ev = (atom < NA) ? __expf(-diff * diff) : 0.0f;
                aFrag[chunk][flane][e] = (_Float16)Ev;
            }
        }
        __syncthreads();

        v8f acc1 = {}, acc2 = {};
        #pragma unroll
        for (int cc = 0; cc < 4; ++cc) {
            v16h aF = *(const v16h*)(&aFrag[cc][lane][0]);
            acc1 = __builtin_amdgcn_wmma_f32_16x16x32_f16(false, aF, false, bW1[cc],
                                                          (short)0, acc1, false, false);
            acc2 = __builtin_amdgcn_wmma_f32_16x16x32_f16(false, aF, false, bW2[cc],
                                                          (short)0, acc2, false, false);
        }

        // Weighted atom reduction directly on accumulator layout:
        // lane -> N=fcol; VGPR r -> M = r + 8*h2 (atom row within tile).
        #pragma unroll
        for (int r = 0; r < 8; ++r) {
            int m = r + 8 * h2;
            int atom = a0 + m;
            if (atom < NA) {
                size_t pidx = ((size_t)b * NA + atom) * FD + fcol;
                float w1 = acc1[r] + bf1;
                float w2 = acc2[r] + bf2;
                sH += w1 * phi1[pidx];
                float t2 = w2 * phi2[pidx];
                sVx += t2 * uvec[m][0];
                sVy += t2 * uvec[m][1];
                sVz += t2 * uvec[m][2];
            }
        }
        __syncthreads();                        // before overwriting LDS
    }

    // Lanes L and L+16 hold the same f column (different atom rows): combine.
    sH  += __shfl_down(sH, 16);
    sVx += __shfl_down(sVx, 16);
    sVy += __shfl_down(sVy, 16);
    sVz += __shfl_down(sVz, 16);
    if (h2 == 0) {
        size_t o0 = ((size_t)b * NC + c) * FD + fcol;
        out[o0] = Hc[o0] + sH;                  // H + dH
        float* dV = out + (size_t)B_DIM * NC * FD;
        size_t o1 = o0 * 3;
        dV[o1 + 0] = sVx;
        dV[o1 + 1] = sVy;
        dV[o1 + 2] = sVz;
    }
}

extern "C" void kernel_launch(void* const* d_in, const int* in_sizes, int n_in,
                              void* d_out, int out_size, void* d_ws, size_t ws_size,
                              hipStream_t stream) {
    // setup_inputs order: 0 assign(unused), 1 h, 2 H, 3 cg_xyz, 4 xyz,
    // 5 cg_adj(unused), 6 Wf, 7 bf, 8 W1, 9 b1, 10 W2, 11 b2
    const float* hin = (const float*)d_in[1];
    const float* Hc  = (const float*)d_in[2];
    const float* cg  = (const float*)d_in[3];
    const float* xyz = (const float*)d_in[4];
    const float* Wf  = (const float*)d_in[6];
    const float* bf  = (const float*)d_in[7];
    const float* W1  = (const float*)d_in[8];
    const float* b1  = (const float*)d_in[9];
    const float* W2  = (const float*)d_in[10];
    const float* b2  = (const float*)d_in[11];
    float* out = (float*)d_out;

    float* phi1 = (float*)d_ws;                       // [B, NA, F] f32
    float* phi2 = phi1 + (size_t)B_DIM * NA * FD;     // [B, NA, F] f32

    phi_kernel<<<B_DIM * NTILE, 256, 0, stream>>>(hin, W1, b1, W2, b2, phi1, phi2);
    agg_kernel<<<B_DIM * NC, 256, 0, stream>>>(Hc, cg, xyz, Wf, bf, phi1, phi2, out);
}
